// Seq2Seq_6957847020228
// MI455X (gfx1250) — matrix operand order repaired and checked by reading, not compiled, not run on hardware
//
#include <hip/hip_runtime.h>
#include <math.h>

typedef unsigned short u16;                       // bf16 storage
typedef __bf16 bf16_t;
typedef bf16_t bf16x8  __attribute__((ext_vector_type(8)));
typedef bf16_t bf16x16 __attribute__((ext_vector_type(16)));
typedef float  f32x8   __attribute__((ext_vector_type(8)));

#define B_   32
#define S_   128
#define E_   512
#define H_   1024
#define H2_  2048
#define H3_  3072
#define H4_  4096
#define V1_  31999
#define VP_  32000
#define T_   50

__device__ __forceinline__ u16 f2bf(float f) {
  unsigned u = __float_as_uint(f);
  u += 0x7FFFu + ((u >> 16) & 1u);                // round-to-nearest-even
  return (u16)(u >> 16);
}
__device__ __forceinline__ float sigf(float x) { return 1.0f / (1.0f + expf(-x)); }

// ---------------------------------------------------------------------------
// Pack fp32 weight W[N,K] (row-major) into WMMA B-fragment tiles (bf16).
// Tile (nt,kt) -> 32 lanes x 16 bf16 contiguous; lane<16: K=kt*32+j, col=lane;
// lane>=16: K=kt*32+16+j, col=lane-16.  Columns >= N are zero-padded.
// ---------------------------------------------------------------------------
__global__ __launch_bounds__(256) void pack_b_kernel(
    const float* __restrict__ W, u16* __restrict__ out, int N, int K, int Np) {
  size_t idx = (size_t)blockIdx.x * blockDim.x + threadIdx.x;
  size_t total = (size_t)Np * K;
  if (idx >= total) return;
  int    j    = (int)(idx & 15);
  int    lane = (int)((idx >> 4) & 31);
  size_t tile = idx >> 9;
  int    Kt   = K >> 5;
  int    nt   = (int)(tile / (size_t)Kt);
  int    kt   = (int)(tile - (size_t)nt * Kt);
  int    n    = (nt << 4) + (lane & 15);
  int    k    = (kt << 5) + ((lane >> 4) << 4) + j;
  float  v    = (n < N) ? W[(size_t)n * K + k] : 0.0f;
  out[idx] = f2bf(v);
}

// ---------------------------------------------------------------------------
// Generic WMMA GEMM with 2x2 register blocking (32x32 output per wave):
//   Cout[M,N] = A[M,K](bf16,row-major) * Bpacked + (Cin | bias | 0)
// 4 independent v_wmma_f32_16x16x32_bf16 per K-step, sharing 2 A + 2 B frags.
// Requires M % 32 == 0 and Np % 32 == 0 (true for all call sites here).
// ---------------------------------------------------------------------------
__global__ __launch_bounds__(256) void gemm_bf16_wmma(
    const u16* __restrict__ A, int lda,
    const u16* __restrict__ Bp,
    const float* __restrict__ bias,
    const float* __restrict__ Cin, int ldci,
    float* __restrict__ Cout, int ldco,
    u16* __restrict__ Cbf, int ldcb,
    int M, int Np, int K, int Nstore) {
  const int lane = threadIdx.x & 31;
  const int w    = blockIdx.x * (blockDim.x >> 5) + (threadIdx.x >> 5);
  const int Mt2 = M >> 5, Nt2 = Np >> 5, Kt = K >> 5;
  if (w >= Mt2 * Nt2) return;                      // wave-uniform
  const int mt2 = w / Nt2, nt2 = w - mt2 * Nt2;

  const int nCol0 = (nt2 << 5) + (lane & 15);      // C column of n-tile 0
  const int mRow0 = (mt2 << 5) + ((lane >> 4) << 3); // first row of m-tile 0

  f32x8 acc[2][2];                                 // [mi][ni]
#pragma unroll
  for (int mi = 0; mi < 2; ++mi)
#pragma unroll
    for (int ni = 0; ni < 2; ++ni) {
      const int nC = nCol0 + ni * 16;
      const int mR = mRow0 + mi * 16;
#pragma unroll
      for (int r = 0; r < 8; ++r) {
        float v = 0.0f;
        if (Cin)                        v = Cin[(size_t)(mR + r) * ldci + nC];
        else if (bias && nC < Nstore)   v = bias[nC];
        acc[mi][ni][r] = v;
      }
    }

  // A fragment (contiguous K to match packed B): lane<16 -> K 0..15; lane>=16 -> K 16..31
  const int  mA   = (mt2 << 5) + (lane & 15);
  const int  kSub = (lane >> 4) << 4;
  const u16* aRow0 = A + (size_t)mA * lda + kSub;          // m-tile 0
  const u16* aRow1 = aRow0 + (size_t)16 * lda;             // m-tile 1
  const u16* bPtr0 = Bp + (size_t)(nt2 * 2) * Kt * 512 + lane * 16;
  const u16* bPtr1 = bPtr0 + (size_t)Kt * 512;

  for (int kt = 0; kt < Kt; ++kt) {
    bf16x8 a00 = *reinterpret_cast<const bf16x8*>(aRow0);
    bf16x8 a01 = *reinterpret_cast<const bf16x8*>(aRow0 + 8);
    bf16x8 a10 = *reinterpret_cast<const bf16x8*>(aRow1);
    bf16x8 a11 = *reinterpret_cast<const bf16x8*>(aRow1 + 8);
    bf16x16 av0, av1;
#pragma unroll
    for (int i = 0; i < 8; ++i) {
      av0[i] = a00[i]; av0[i + 8] = a01[i];
      av1[i] = a10[i]; av1[i + 8] = a11[i];
    }
    bf16x16 bv0 = *reinterpret_cast<const bf16x16*>(bPtr0);
    bf16x16 bv1 = *reinterpret_cast<const bf16x16*>(bPtr1);
    acc[0][0] = __builtin_amdgcn_wmma_f32_16x16x32_bf16(false, av0, false, bv0,
                                                        (short)0, acc[0][0], false, false);
    acc[0][1] = __builtin_amdgcn_wmma_f32_16x16x32_bf16(false, av0, false, bv1,
                                                        (short)0, acc[0][1], false, false);
    acc[1][0] = __builtin_amdgcn_wmma_f32_16x16x32_bf16(false, av1, false, bv0,
                                                        (short)0, acc[1][0], false, false);
    acc[1][1] = __builtin_amdgcn_wmma_f32_16x16x32_bf16(false, av1, false, bv1,
                                                        (short)0, acc[1][1], false, false);
    aRow0 += 32; aRow1 += 32;
    bPtr0 += 512; bPtr1 += 512;
  }

#pragma unroll
  for (int mi = 0; mi < 2; ++mi)
#pragma unroll
    for (int ni = 0; ni < 2; ++ni) {
      const int nC = nCol0 + ni * 16;
      if (nC < Nstore) {
        const int mR = mRow0 + mi * 16;
#pragma unroll
        for (int r = 0; r < 8; ++r) {
          float v = acc[mi][ni][r];
          Cout[(size_t)(mR + r) * ldco + nC] = v;
          if (Cbf) Cbf[(size_t)(mR + r) * ldcb + nC] = f2bf(v);
        }
      }
    }
}

// ---------------------------------------------------------------------------
// Embedding gathers (fp32 table -> bf16 activations)
// ---------------------------------------------------------------------------
__global__ void embed_enc_kernel(const int* __restrict__ inputs,
                                 const float* __restrict__ emb,
                                 u16* __restrict__ embT) {        // [S,B,E]
  size_t idx = (size_t)blockIdx.x * blockDim.x + threadIdx.x;
  if (idx >= (size_t)S_ * B_ * E_) return;
  int e  = (int)(idx % E_);
  int sb = (int)(idx / E_);
  int b  = sb % B_;
  int s  = sb / B_;
  int tok = inputs[b * S_ + s];
  embT[idx] = f2bf(emb[(size_t)tok * E_ + e]);
}

__global__ void embed_dec_kernel(const int* __restrict__ tok,
                                 const float* __restrict__ emb,
                                 u16* __restrict__ x) {           // [B,E]
  int idx = blockIdx.x * blockDim.x + threadIdx.x;
  if (idx >= B_ * E_) return;
  int e = idx % E_, b = idx / E_;
  x[idx] = f2bf(emb[(size_t)tok[b] * E_ + e]);
}

// ---------------------------------------------------------------------------
// LSTM pointwise cell: gates G[B,4H] (i,f,g,o) -> h,c ; optional seq output
// ---------------------------------------------------------------------------
__global__ void lstm_cell_kernel(const float* __restrict__ G,
                                 float* __restrict__ h, float* __restrict__ c,
                                 u16* __restrict__ hbf,
                                 float* __restrict__ yf, u16* __restrict__ yb,
                                 long long yoff, int ystride) {
  int idx = blockIdx.x * blockDim.x + threadIdx.x;
  if (idx >= B_ * H_) return;
  int j = idx % H_, b = idx / H_;
  const float* g = G + (size_t)b * H4_;
  float ig = sigf(g[j]);
  float fg = sigf(g[H_ + j]);
  float gg = tanhf(g[2 * H_ + j]);
  float og = sigf(g[3 * H_ + j]);
  float cn = fg * c[idx] + ig * gg;
  float hn = og * tanhf(cn);
  c[idx] = cn; h[idx] = hn; hbf[idx] = f2bf(hn);
  if (yf) yf[yoff + (size_t)b * ystride + j] = hn;
  if (yb) yb[yoff + (size_t)b * ystride + j] = f2bf(hn);
}

// Sum the 4 encoder final states; init both decoder layers + START tokens.
__global__ void init_decoder_kernel(const float* h0f, const float* h0b,
                                    const float* h1f, const float* h1b,
                                    const float* c0f, const float* c0b,
                                    const float* c1f, const float* c1b,
                                    float* dh0, float* dc0, float* dh1, float* dc1,
                                    u16* dh0b, u16* dh1b, int* tok) {
  int idx = blockIdx.x * blockDim.x + threadIdx.x;
  if (idx >= B_ * H_) return;
  float hs = h0f[idx] + h0b[idx] + h1f[idx] + h1b[idx];
  float cs = c0f[idx] + c0b[idx] + c1f[idx] + c1b[idx];
  dh0[idx] = hs; dh1[idx] = hs; dc0[idx] = cs; dc1[idx] = cs;
  u16 hb = f2bf(hs); dh0b[idx] = hb; dh1b[idx] = hb;
  if (idx < B_) tok[idx] = 1;                      // START
}

// ---------------------------------------------------------------------------
// Attention: scores->softmax->context, emits merge input [h1 | ctx] as bf16.
// One block per batch row; enc layout [S,B,2H].
// ---------------------------------------------------------------------------
__global__ __launch_bounds__(256) void attention_kernel(
    const float* __restrict__ q, const float* __restrict__ enc,
    const int* __restrict__ inputs, const float* __restrict__ h1,
    u16* __restrict__ cat) {
  int b = blockIdx.x;
  __shared__ float wsm[S_];
  int lane = threadIdx.x & 31, wv = threadIdx.x >> 5;
  for (int j = wv; j < S_; j += 8) {
    float p = 0.0f;
    const float* er = enc + ((size_t)j * B_ + b) * H2_;
    const float* qr = q + (size_t)b * H2_;
    for (int d = lane; d < H2_; d += 32) p += qr[d] * er[d];
    for (int off = 16; off > 0; off >>= 1) p += __shfl_down(p, off, 32);
    if (lane == 0) wsm[j] = (inputs[b * S_ + j] == 0) ? -1e20f : p;
  }
  __syncthreads();
  if (threadIdx.x == 0) {
    float mx = -INFINITY;
    for (int j = 0; j < S_; ++j) mx = fmaxf(mx, wsm[j]);
    float sum = 0.0f;
    for (int j = 0; j < S_; ++j) { float e = expf(wsm[j] - mx); wsm[j] = e; sum += e; }
    float inv = 1.0f / sum;
    for (int j = 0; j < S_; ++j) wsm[j] *= inv;
  }
  __syncthreads();
  for (int jj = threadIdx.x; jj < H_; jj += 256)
    cat[(size_t)b * H3_ + jj] = f2bf(h1[(size_t)b * H_ + jj]);
  for (int d = threadIdx.x; d < H2_; d += 256) {
    float acc = 0.0f;
    for (int j = 0; j < S_; ++j) acc += wsm[j] * enc[((size_t)j * B_ + b) * H2_ + d];
    cat[(size_t)b * H3_ + H_ + d] = f2bf(acc);
  }
}

// ---------------------------------------------------------------------------
// log_softmax over 31999 logits + argmax (first-max tie-break) per batch row.
// ---------------------------------------------------------------------------
__global__ __launch_bounds__(256) void lsm_argmax_kernel(
    const float* __restrict__ logits, float* __restrict__ scores,
    float* __restrict__ preds, int* __restrict__ tok, int t) {
  int b = blockIdx.x;
  const float* row = logits + (size_t)b * V1_;
  __shared__ float smax[256]; __shared__ int sidx[256]; __shared__ float ssum[256];
  float mx = -INFINITY; int am = 0;
  for (int v = threadIdx.x; v < V1_; v += 256) {
    float x = row[v];
    if (x > mx) { mx = x; am = v; }
  }
  smax[threadIdx.x] = mx; sidx[threadIdx.x] = am;
  __syncthreads();
  if (threadIdx.x == 0) {
    for (int i = 1; i < 256; ++i)
      if (smax[i] > smax[0] || (smax[i] == smax[0] && sidx[i] < sidx[0])) {
        smax[0] = smax[i]; sidx[0] = sidx[i];
      }
  }
  __syncthreads();
  mx = smax[0];
  float s = 0.0f;
  for (int v = threadIdx.x; v < V1_; v += 256) s += expf(row[v] - mx);
  ssum[threadIdx.x] = s;
  __syncthreads();
  if (threadIdx.x == 0) {
    float tot = 0.0f;
    for (int i = 0; i < 256; ++i) tot += ssum[i];
    ssum[0] = logf(tot) + mx;
  }
  __syncthreads();
  float lse = ssum[0];
  float* srow = scores + ((size_t)b * T_ + t) * V1_;
  for (int v = threadIdx.x; v < V1_; v += 256) srow[v] = row[v] - lse;
  if (threadIdx.x == 0) {
    preds[(size_t)b * T_ + t] = (float)sidx[0];
    tok[b] = sidx[0];
  }
}

// ===========================================================================
extern "C" void kernel_launch(void* const* d_in, const int* in_sizes, int n_in,
                              void* d_out, int out_size, void* d_ws, size_t ws_size,
                              hipStream_t stream) {
  const int*   inputs  = (const int*)  d_in[0];
  const float* enc_emb = (const float*)d_in[1];
  const float* l0f_wih = (const float*)d_in[2];
  const float* l0f_whh = (const float*)d_in[3];
  const float* l0f_b   = (const float*)d_in[4];
  const float* l0b_wih = (const float*)d_in[5];
  const float* l0b_whh = (const float*)d_in[6];
  const float* l0b_b   = (const float*)d_in[7];
  const float* l1f_wih = (const float*)d_in[8];
  const float* l1f_whh = (const float*)d_in[9];
  const float* l1f_b   = (const float*)d_in[10];
  const float* l1b_wih = (const float*)d_in[11];
  const float* l1b_whh = (const float*)d_in[12];
  const float* l1b_b   = (const float*)d_in[13];
  const float* dec_emb = (const float*)d_in[14];
  const float* d0_wih  = (const float*)d_in[15];
  const float* d0_whh  = (const float*)d_in[16];
  const float* d0_b    = (const float*)d_in[17];
  const float* d1_wih  = (const float*)d_in[18];
  const float* d1_whh  = (const float*)d_in[19];
  const float* d1_b    = (const float*)d_in[20];
  const float* attn_w  = (const float*)d_in[21];
  const float* merge_w = (const float*)d_in[22];
  const float* out_w   = (const float*)d_in[23];
  const float* out_b   = (const float*)d_in[24];

  char* base = (char*)d_ws;
  size_t off = 0;
  auto alloc = [&](size_t bytes) -> void* {
    void* p = base + off;
    off += (bytes + 255) & ~(size_t)255;
    return p;
  };

  // Packed bf16 weights (B-fragment tile layout)
  u16* p_l0f_wih = (u16*)alloc((size_t)E_  * H4_ * 2);
  u16* p_l0f_whh = (u16*)alloc((size_t)H_  * H4_ * 2);
  u16* p_l0b_wih = (u16*)alloc((size_t)E_  * H4_ * 2);
  u16* p_l0b_whh = (u16*)alloc((size_t)H_  * H4_ * 2);
  u16* p_l1f_wih = (u16*)alloc((size_t)H2_ * H4_ * 2);
  u16* p_l1f_whh = (u16*)alloc((size_t)H_  * H4_ * 2);
  u16* p_l1b_wih = (u16*)alloc((size_t)H2_ * H4_ * 2);
  u16* p_l1b_whh = (u16*)alloc((size_t)H_  * H4_ * 2);
  u16* p_d0_wih  = (u16*)alloc((size_t)E_  * H4_ * 2);
  u16* p_d0_whh  = (u16*)alloc((size_t)H_  * H4_ * 2);
  u16* p_d1_wih  = (u16*)alloc((size_t)H_  * H4_ * 2);
  u16* p_d1_whh  = (u16*)alloc((size_t)H_  * H4_ * 2);
  u16* p_attn    = (u16*)alloc((size_t)H_  * H2_ * 2);
  u16* p_merge   = (u16*)alloc((size_t)H3_ * H_  * 2);
  u16* p_outw    = (u16*)alloc((size_t)H_  * VP_ * 2);

  // Activations / scratch
  u16*   embT    = (u16*)  alloc((size_t)S_ * B_ * E_  * 2);   // [S,B,E]
  float* xproj   = (float*)alloc((size_t)S_ * B_ * H4_ * 4);   // [S,B,4H] (reused x4)
  u16*   y0      = (u16*)  alloc((size_t)S_ * B_ * H2_ * 2);   // [S,B,2H] bf16
  float* enc_out = (float*)alloc((size_t)S_ * B_ * H2_ * 4);   // [S,B,2H] fp32
  float* G       = (float*)alloc((size_t)B_ * H4_ * 4);        // gate buffer

  float* eh[4]; float* ec[4]; u16* ehb[4];
  for (int i = 0; i < 4; ++i) {
    eh[i]  = (float*)alloc((size_t)B_ * H_ * 4);
    ec[i]  = (float*)alloc((size_t)B_ * H_ * 4);
    ehb[i] = (u16*)  alloc((size_t)B_ * H_ * 2);
  }
  float* dh0  = (float*)alloc((size_t)B_ * H_ * 4);
  float* dc0  = (float*)alloc((size_t)B_ * H_ * 4);
  float* dh1  = (float*)alloc((size_t)B_ * H_ * 4);
  float* dc1  = (float*)alloc((size_t)B_ * H_ * 4);
  u16*   dh0b = (u16*)  alloc((size_t)B_ * H_ * 2);
  u16*   dh1b = (u16*)  alloc((size_t)B_ * H_ * 2);
  u16*   xdec = (u16*)  alloc((size_t)B_ * E_ * 2);
  float* qbuf = (float*)alloc((size_t)B_ * H2_ * 4);
  u16*   cat  = (u16*)  alloc((size_t)B_ * H3_ * 2);
  float* mo   = (float*)alloc((size_t)B_ * H_ * 4);
  u16*   mob  = (u16*)  alloc((size_t)B_ * H_ * 2);
  float* lgt  = (float*)alloc((size_t)B_ * V1_ * 4);
  int*   tok  = (int*)  alloc((size_t)B_ * 4);

  auto pack = [&](const float* W, u16* dst, int N, int K, int Np) {
    size_t total = (size_t)Np * K;
    pack_b_kernel<<<(int)((total + 255) / 256), 256, 0, stream>>>(W, dst, N, K, Np);
  };
  auto gemm = [&](const u16* A, int lda, const u16* Bp, const float* bias,
                  const float* Cin, int ldci, float* Cout, int ldco,
                  u16* Cbf, int ldcb, int M, int Np, int K, int Nstore) {
    int tiles  = (M >> 5) * (Np >> 5);             // 32x32 per wave
    int blocks = (tiles + 7) >> 3;                 // 8 waves / block
    gemm_bf16_wmma<<<blocks, 256, 0, stream>>>(A, lda, Bp, bias, Cin, ldci,
                                               Cout, ldco, Cbf, ldcb,
                                               M, Np, K, Nstore);
  };

  // ---- Pack all weights (once per call) ----
  pack(l0f_wih, p_l0f_wih, H4_, E_,  H4_); pack(l0f_whh, p_l0f_whh, H4_, H_,  H4_);
  pack(l0b_wih, p_l0b_wih, H4_, E_,  H4_); pack(l0b_whh, p_l0b_whh, H4_, H_,  H4_);
  pack(l1f_wih, p_l1f_wih, H4_, H2_, H4_); pack(l1f_whh, p_l1f_whh, H4_, H_,  H4_);
  pack(l1b_wih, p_l1b_wih, H4_, H2_, H4_); pack(l1b_whh, p_l1b_whh, H4_, H_,  H4_);
  pack(d0_wih,  p_d0_wih,  H4_, E_,  H4_); pack(d0_whh,  p_d0_whh,  H4_, H_,  H4_);
  pack(d1_wih,  p_d1_wih,  H4_, H_,  H4_); pack(d1_whh,  p_d1_whh,  H4_, H_,  H4_);
  pack(attn_w,  p_attn,    H2_, H_,  H2_);
  pack(merge_w, p_merge,   H_,  H3_, H_);
  pack(out_w,   p_outw,    V1_, H_,  VP_);

  // ---- Encoder ----
  embed_enc_kernel<<<(S_ * B_ * E_ + 255) / 256, 256, 0, stream>>>(inputs, enc_emb, embT);

  auto run_lstm = [&](const u16* Ain, int lda, int K, const u16* wihp, const u16* whhp,
                      const float* bias, bool rev, float* h, float* c, u16* hb,
                      float* yf, u16* yb, int dirOff) {
    // Hoisted input projection for all S timesteps (big WMMA GEMM, bias folded)
    gemm(Ain, lda, wihp, bias, nullptr, 0, xproj, H4_, nullptr, 0,
         S_ * B_, H4_, K, H4_);
    hipMemsetAsync(h,  0, (size_t)B_ * H_ * 4, stream);
    hipMemsetAsync(c,  0, (size_t)B_ * H_ * 4, stream);
    hipMemsetAsync(hb, 0, (size_t)B_ * H_ * 2, stream);
    for (int i = 0; i < S_; ++i) {
      int t = rev ? (S_ - 1 - i) : i;
      const float* Cin = xproj + (size_t)t * B_ * H4_;
      gemm(hb, H_, whhp, nullptr, Cin, H4_, G, H4_, nullptr, 0, B_, H4_, H_, H4_);
      long long yoff = (long long)t * B_ * H2_ + dirOff;
      lstm_cell_kernel<<<(B_ * H_ + 255) / 256, 256, 0, stream>>>(
          G, h, c, hb, yf, yb, yoff, H2_);
    }
  };
  run_lstm(embT, E_,  E_,  p_l0f_wih, p_l0f_whh, l0f_b, false,
           eh[0], ec[0], ehb[0], nullptr, y0, 0);
  run_lstm(embT, E_,  E_,  p_l0b_wih, p_l0b_whh, l0b_b, true,
           eh[1], ec[1], ehb[1], nullptr, y0, H_);
  run_lstm(y0,   H2_, H2_, p_l1f_wih, p_l1f_whh, l1f_b, false,
           eh[2], ec[2], ehb[2], enc_out, nullptr, 0);
  run_lstm(y0,   H2_, H2_, p_l1b_wih, p_l1b_whh, l1b_b, true,
           eh[3], ec[3], ehb[3], enc_out, nullptr, H_);

  // ---- Decoder init (h_sum/c_sum to both layers, START tokens) ----
  init_decoder_kernel<<<(B_ * H_ + 255) / 256, 256, 0, stream>>>(
      eh[0], eh[1], eh[2], eh[3], ec[0], ec[1], ec[2], ec[3],
      dh0, dc0, dh1, dc1, dh0b, dh1b, tok);

  float* scores = (float*)d_out;
  float* preds  = scores + (size_t)B_ * T_ * V1_;

  // ---- Greedy decode loop ----
  for (int t = 0; t < T_; ++t) {
    embed_dec_kernel<<<(B_ * E_ + 255) / 256, 256, 0, stream>>>(tok, dec_emb, xdec);
    // layer 0
    gemm(xdec, E_, p_d0_wih, d0_b, nullptr, 0, G, H4_, nullptr, 0, B_, H4_, E_, H4_);
    gemm(dh0b, H_, p_d0_whh, nullptr, G, H4_, G, H4_, nullptr, 0, B_, H4_, H_, H4_);
    lstm_cell_kernel<<<(B_ * H_ + 255) / 256, 256, 0, stream>>>(
        G, dh0, dc0, dh0b, nullptr, nullptr, 0, 0);
    // layer 1
    gemm(dh0b, H_, p_d1_wih, d1_b, nullptr, 0, G, H4_, nullptr, 0, B_, H4_, H_, H4_);
    gemm(dh1b, H_, p_d1_whh, nullptr, G, H4_, G, H4_, nullptr, 0, B_, H4_, H_, H4_);
    lstm_cell_kernel<<<(B_ * H_ + 255) / 256, 256, 0, stream>>>(
        G, dh1, dc1, dh1b, nullptr, nullptr, 0, 0);
    // attention: q = h1 @ attn_w.T ; softmax over enc_out ; cat = [h1|ctx]
    gemm(dh1b, H_, p_attn, nullptr, nullptr, 0, qbuf, H2_, nullptr, 0,
         B_, H2_, H_, H2_);
    attention_kernel<<<B_, 256, 0, stream>>>(qbuf, enc_out, inputs, dh1, cat);
    // merge + output projection (N padded to 32000, store 31999)
    gemm(cat, H3_, p_merge, nullptr, nullptr, 0, mo, H_, mob, H_, B_, H_, H3_, H_);
    gemm(mob, H_, p_outw, out_b, nullptr, 0, lgt, V1_, nullptr, 0, B_, VP_, H_, V1_);
    lsm_argmax_kernel<<<B_, 256, 0, stream>>>(lgt, scores, preds, tok, t);
  }
}
